// MAB_22436909154335
// MI455X (gfx1250) — compile-verified, matrix operand order checked by reading, not run on hardware
//
#include <hip/hip_runtime.h>
#include <hip/hip_bf16.h>

typedef __bf16 bf16_t;
typedef __attribute__((ext_vector_type(16))) __bf16 v16bf;
typedef __attribute__((ext_vector_type(8)))  float  v8f;

// ---------------------------------------------------------------------------
// WMMA helper: D = A(16x32 bf16) * B(32x16 bf16) + C(16x16 f32)
// ---------------------------------------------------------------------------
__device__ __forceinline__ v8f wmma_bf16(v16bf a, v16bf b, v8f c) {
  return __builtin_amdgcn_wmma_f32_16x16x32_bf16(
      /*neg_a=*/false, a, /*neg_b=*/false, b,
      /*c_mod=*/(short)0, c, /*reuse_a=*/false, /*reuse_b=*/false);
}

// A/B operand fragment load (16-bit, K=32 slice), per ISA 7.12.2:
// lane L holds row/col (L&15); halfs 0..7 -> K = (L>>4)*8 + i,
// halfs 8..15 -> K = 16 + (L>>4)*8 + (i-8).  Two contiguous 16B loads.
__device__ __forceinline__ v16bf load_ab_bf16(const bf16_t* rowk, int grp) {
  union { uint4 u[2]; v16bf v; } r;
  const uint4* p = reinterpret_cast<const uint4*>(rowk);
  r.u[0] = p[grp];
  r.u[1] = p[2 + grp];
  return r.v;
}

__device__ __forceinline__ v16bf load_ab_f32(const float* rowk, int grp) {
  const float4* p = reinterpret_cast<const float4*>(rowk);
  float4 x0 = p[2 * grp], x1 = p[2 * grp + 1];
  float4 x2 = p[4 + 2 * grp], x3 = p[4 + 2 * grp + 1];
  v16bf r;
  r[0]  = (__bf16)x0.x; r[1]  = (__bf16)x0.y; r[2]  = (__bf16)x0.z; r[3]  = (__bf16)x0.w;
  r[4]  = (__bf16)x1.x; r[5]  = (__bf16)x1.y; r[6]  = (__bf16)x1.z; r[7]  = (__bf16)x1.w;
  r[8]  = (__bf16)x2.x; r[9]  = (__bf16)x2.y; r[10] = (__bf16)x2.z; r[11] = (__bf16)x2.w;
  r[12] = (__bf16)x3.x; r[13] = (__bf16)x3.y; r[14] = (__bf16)x3.z; r[15] = (__bf16)x3.w;
  return r;
}

// ---------------------------------------------------------------------------
// CDNA5 async global->LDS DMA (ASYNCcnt path), ISA 10/15.18.
// vdst = per-lane LDS byte offset, vaddr = per-lane 64-bit global address.
// ---------------------------------------------------------------------------
__device__ __forceinline__ void async_copy_b128(unsigned lds_off, const void* gptr) {
  asm volatile("global_load_async_to_lds_b128 %0, %1, off"
               :: "v"(lds_off), "v"((unsigned long long)(uintptr_t)gptr)
               : "memory");
}
__device__ __forceinline__ void wait_asynccnt0() {
  asm volatile("s_wait_asynccnt 0" ::: "memory");
}
__device__ __forceinline__ unsigned lds_off_u32(const void* p) {
  // generic pointers to LDS carry the 32-bit LDS offset in the low DWORD
  return (unsigned)(uintptr_t)p;
}

// ---------------------------------------------------------------------------
// fp32 -> bf16 convert (weights)
// ---------------------------------------------------------------------------
__global__ void cvt_bf16_kernel(const float* __restrict__ src,
                                bf16_t* __restrict__ dst, int n) {
  int i = blockIdx.x * blockDim.x + threadIdx.x;
  if (i < n) dst[i] = (bf16_t)src[i];
}

// ---------------------------------------------------------------------------
// Projection: P = X @ W^T + b.  X:[rows,256] f32, W:[256out,256in] bf16.
// One wave: 16 rows x 64 cols, K-loop of 32.  mode: 0=Q (f32 full + bf16 head),
// 1=K (bf16 head [B][H][N][32]), 2=V (bf16 transposed [B][H][32][N]).
// ---------------------------------------------------------------------------
__global__ void proj_kernel(const float* __restrict__ X,
                            const bf16_t* __restrict__ Wb,
                            const float* __restrict__ bias,
                            float* __restrict__ outF,
                            bf16_t* __restrict__ outH,
                            int N /*rows per batch*/, int nshift /*log2 N*/,
                            int mode) {
  const int lane = threadIdx.x;
  const int grp  = lane >> 4;
  const int ln   = lane & 15;
  const int row0 = blockIdx.x * 16;
  const int col0 = blockIdx.y * 64;

  v8f acc[4] = {};
  const float* xrow = X + (size_t)(row0 + ln) * 256;
  for (int kk = 0; kk < 256; kk += 32) {
    v16bf a = load_ab_f32(xrow + kk, grp);
#pragma unroll
    for (int t = 0; t < 4; ++t) {
      v16bf b = load_ab_bf16(Wb + (size_t)(col0 + t * 16 + ln) * 256 + kk, grp);
      acc[t] = wmma_bf16(a, b, acc[t]);
    }
  }

  const int bidx = row0 >> nshift;      // whole tile in one batch (N % 16 == 0)
#pragma unroll
  for (int t = 0; t < 4; ++t) {
    const int col = col0 + t * 16 + ln;
    const float bv = bias[col];
    const int h = col >> 5, d = col & 31;
#pragma unroll
    for (int j = 0; j < 8; ++j) {
      const int row = row0 + j + grp * 8;   // C layout: M = j + 8*(lane>=16)
      const int n   = row - (bidx << nshift);
      const float v = acc[t][j] + bv;
      if (mode == 0) {
        outF[(size_t)row * 256 + col] = v;
        outH[(((size_t)bidx * 8 + h) * N + n) * 32 + d] = (bf16_t)v;
      } else if (mode == 1) {
        outH[(((size_t)bidx * 8 + h) * N + n) * 32 + d] = (bf16_t)v;
      } else {
        outH[(((size_t)bidx * 8 + h) * 32 + d) * (size_t)N + n] = (bf16_t)v;
      }
    }
  }
}

// ---------------------------------------------------------------------------
// Flash attention + per-head residual.
// Block = 4 waves = 4 consecutive 16-query tiles sharing one (b,h).
// K/V 32-key chunks are staged into double-buffered LDS with async DMA and
// shared by all 4 waves; DMA of chunk c+1 overlaps compute of chunk c.
// Qb:[B][H][Nq][32], Kb:[B][H][Nk][32], Vt:[B][H][32][Nk], Qf/O:[B][Nq][256]
// ---------------------------------------------------------------------------
__global__ void attn_kernel(const bf16_t* __restrict__ Qb,
                            const bf16_t* __restrict__ Kb,
                            const bf16_t* __restrict__ Vt,
                            const float*  __restrict__ Qf,
                            float* __restrict__ O,
                            int Nq, int Nk) {
  __shared__ __align__(16) bf16_t ldsK[2][32 * 32];  // [keys][hd]
  __shared__ __align__(16) bf16_t ldsV[2][32 * 32];  // [hd][keys]
  __shared__ __align__(16) bf16_t ldsP[4][16 * 32];  // per-wave P tile

  const int tid  = threadIdx.x;        // 0..127
  const int wave = tid >> 5;
  const int lane = tid & 31;
  const int grp  = lane >> 4;
  const int ln   = lane & 15;
  const int q0   = (blockIdx.x * 4 + wave) * 16;
  const int h    = blockIdx.y;
  const int b    = blockIdx.z;
  const size_t bh = (size_t)b * 8 + h;

  const bf16_t* Kbase = Kb + bh * (size_t)Nk * 32;
  const bf16_t* Vbase = Vt + bh * 32 * (size_t)Nk;

  // Per-thread DMA assignment: 32 rows x 4 x 16B per tile, 128 threads.
  const int drow  = tid >> 2;          // 0..31
  const int dpart = tid & 3;           // 0..3 (16B units)
  const unsigned kOff0 = lds_off_u32(&ldsK[0][0]) + drow * 64 + dpart * 16;
  const unsigned kOff1 = lds_off_u32(&ldsK[1][0]) + drow * 64 + dpart * 16;
  const unsigned vOff0 = lds_off_u32(&ldsV[0][0]) + drow * 64 + dpart * 16;
  const unsigned vOff1 = lds_off_u32(&ldsV[1][0]) + drow * 64 + dpart * 16;

  v16bf aq = load_ab_bf16(Qb + (bh * Nq + q0 + ln) * 32, grp);

  v8f o0 = {}, o1 = {};
  float m[8], l[8];
#pragma unroll
  for (int j = 0; j < 8; ++j) { m[j] = -1e30f; l[j] = 0.f; }
  const float scale = 0.0625f;  // 1/sqrt(DIM_V=256)

  bf16_t* plds = &ldsP[wave][0];

  // prefetch chunk 0 into buffer 0
  async_copy_b128(kOff0, Kbase + (size_t)drow * 32 + dpart * 8);
  async_copy_b128(vOff0, Vbase + (size_t)drow * Nk + dpart * 8);

  const int nchunks = Nk >> 5;
  for (int c = 0; c < nchunks; ++c) {
    const int buf = c & 1;
    // chunk c arrived (this wave's DMA) ...
    wait_asynccnt0();
    // ... visible to all waves, and everyone is done reading the buffer that
    // the next prefetch will overwrite.
    __syncthreads();

    if (c + 1 < nchunks) {
      const int kc1 = (c + 1) << 5;
      const unsigned ko = (buf ? kOff0 : kOff1);
      const unsigned vo = (buf ? vOff0 : vOff1);
      async_copy_b128(ko, Kbase + (size_t)(kc1 + drow) * 32 + dpart * 8);
      async_copy_b128(vo, Vbase + (size_t)drow * Nk + kc1 + dpart * 8);
    }

    const bf16_t* kl = &ldsK[buf][0];
    const bf16_t* vl = &ldsV[buf][0];

    // scores: two 16x16 tiles covering keys kc..kc+31 (from LDS)
    v16bf kb0 = load_ab_bf16(kl + (size_t)ln * 32, grp);
    v16bf kb1 = load_ab_bf16(kl + (size_t)(16 + ln) * 32, grp);
    v8f s0 = {}, s1 = {};
    s0 = wmma_bf16(aq, kb0, s0);
    s1 = wmma_bf16(aq, kb1, s1);

    float p0[8], p1[8];
#pragma unroll
    for (int j = 0; j < 8; ++j) {
      float a0 = s0[j] * scale, a1 = s1[j] * scale;
      float mx = fmaxf(a0, a1);
      mx = fmaxf(mx, __shfl_xor(mx, 1, 16));
      mx = fmaxf(mx, __shfl_xor(mx, 2, 16));
      mx = fmaxf(mx, __shfl_xor(mx, 4, 16));
      mx = fmaxf(mx, __shfl_xor(mx, 8, 16));
      const float mn = fmaxf(m[j], mx);
      const float alpha = __expf(m[j] - mn);
      m[j] = mn;
      p0[j] = __expf(a0 - mn);
      p1[j] = __expf(a1 - mn);
      float rs = p0[j] + p1[j];
      rs += __shfl_xor(rs, 1, 16);
      rs += __shfl_xor(rs, 2, 16);
      rs += __shfl_xor(rs, 4, 16);
      rs += __shfl_xor(rs, 8, 16);
      l[j] = l[j] * alpha + rs;
      o0[j] *= alpha;
      o1[j] *= alpha;
    }

    // C-layout -> A-layout for P via per-wave LDS (in-order per wave)
#pragma unroll
    for (int j = 0; j < 8; ++j) {
      const int row = j + grp * 8;
      plds[row * 32 + ln]      = (bf16_t)p0[j];
      plds[row * 32 + 16 + ln] = (bf16_t)p1[j];
    }
    __builtin_amdgcn_wave_barrier();
    v16bf ap = load_ab_bf16(&plds[ln * 32], grp);
    __builtin_amdgcn_wave_barrier();

    // O += P(16x32) @ V(32x32): two WMMAs (hd cols 0-15, 16-31)
    v16bf vb0 = load_ab_bf16(vl + (size_t)ln * 32, grp);
    v16bf vb1 = load_ab_bf16(vl + (size_t)(16 + ln) * 32, grp);
    o0 = wmma_bf16(ap, vb0, o0);
    o1 = wmma_bf16(ap, vb1, o1);
  }

  // epilogue: O = Qh + softmax-normalized O
#pragma unroll
  for (int j = 0; j < 8; ++j) {
    const int row = q0 + j + grp * 8;
    const float inv = 1.0f / l[j];
    const size_t base = ((size_t)b * Nq + row) * 256;
    const int c0 = h * 32 + ln, c1 = c0 + 16;
    O[base + c0] = Qf[base + c0] + o0[j] * inv;
    O[base + c1] = Qf[base + c1] + o1[j] * inv;
  }
}

// ---------------------------------------------------------------------------
// LayerNorm over 256 features, one wave per row.  Optional bf16 mirror copy.
// ---------------------------------------------------------------------------
__global__ void ln_kernel(const float* __restrict__ X,
                          const float* __restrict__ g,
                          const float* __restrict__ be,
                          float* __restrict__ Yf,
                          bf16_t* __restrict__ Yb, int rows) {
  const int wave = threadIdx.x >> 5;
  const int lane = threadIdx.x & 31;
  const int row = blockIdx.x * (blockDim.x >> 5) + wave;
  if (row >= rows) return;
  const float* x = X + (size_t)row * 256;
  float v[8], s = 0.f;
#pragma unroll
  for (int i = 0; i < 8; ++i) { v[i] = x[lane + 32 * i]; s += v[i]; }
#pragma unroll
  for (int msk = 1; msk < 32; msk <<= 1) s += __shfl_xor(s, msk, 32);
  const float mu = s * (1.f / 256.f);
  float var = 0.f;
#pragma unroll
  for (int i = 0; i < 8; ++i) { const float d = v[i] - mu; var += d * d; }
#pragma unroll
  for (int msk = 1; msk < 32; msk <<= 1) var += __shfl_xor(var, msk, 32);
  const float r = rsqrtf(var * (1.f / 256.f) + 1e-5f);
#pragma unroll
  for (int i = 0; i < 8; ++i) {
    const int c = lane + 32 * i;
    const float y = (v[i] - mu) * r * g[c] + be[c];
    Yf[(size_t)row * 256 + c] = y;
    if (Yb) Yb[(size_t)row * 256 + c] = (bf16_t)y;
  }
}

// ---------------------------------------------------------------------------
// MLP: Z = Xf + relu(Xb @ Wo^T + bo).  Same tiling as proj_kernel.
// ---------------------------------------------------------------------------
__global__ void ffn_kernel(const bf16_t* __restrict__ Xb,
                           const bf16_t* __restrict__ Wb,
                           const float* __restrict__ bias,
                           const float* __restrict__ Xf,
                           float* __restrict__ Z) {
  const int lane = threadIdx.x;
  const int grp  = lane >> 4;
  const int ln   = lane & 15;
  const int row0 = blockIdx.x * 16;
  const int col0 = blockIdx.y * 64;

  v8f acc[4] = {};
  const bf16_t* xrow = Xb + (size_t)(row0 + ln) * 256;
  for (int kk = 0; kk < 256; kk += 32) {
    v16bf a = load_ab_bf16(xrow + kk, grp);
#pragma unroll
    for (int t = 0; t < 4; ++t) {
      v16bf b = load_ab_bf16(Wb + (size_t)(col0 + t * 16 + ln) * 256 + kk, grp);
      acc[t] = wmma_bf16(a, b, acc[t]);
    }
  }
#pragma unroll
  for (int t = 0; t < 4; ++t) {
    const int col = col0 + t * 16 + ln;
    const float bv = bias[col];
#pragma unroll
    for (int j = 0; j < 8; ++j) {
      const int row = row0 + j + grp * 8;
      const size_t idx = (size_t)row * 256 + col;
      Z[idx] = Xf[idx] + fmaxf(acc[t][j] + bv, 0.f);
    }
  }
}

// ---------------------------------------------------------------------------
// Host-side launch
// ---------------------------------------------------------------------------
extern "C" void kernel_launch(void* const* d_in, const int* in_sizes, int n_in,
                              void* d_out, int out_size, void* d_ws, size_t ws_size,
                              hipStream_t stream) {
  const float* q  = (const float*)d_in[0];
  const float* k  = (const float*)d_in[1];
  const float* Wq = (const float*)d_in[2];
  const float* bq = (const float*)d_in[3];
  const float* Wk = (const float*)d_in[4];
  const float* bk = (const float*)d_in[5];
  const float* Wv = (const float*)d_in[6];
  const float* bv = (const float*)d_in[7];
  const float* Wo = (const float*)d_in[8];
  const float* bo = (const float*)d_in[9];
  const float* g0 = (const float*)d_in[10];
  const float* b0 = (const float*)d_in[11];
  const float* g1 = (const float*)d_in[12];
  const float* b1 = (const float*)d_in[13];
  float* out = (float*)d_out;

  const int B     = 4;
  const int rowsQ = in_sizes[0] / 256;     // B * Nq
  const int rowsK = in_sizes[1] / 256;     // B * Nk
  const int Nq    = rowsQ / B;
  const int Nk    = rowsK / B;
  int nqShift = 0; while ((1 << nqShift) < Nq) ++nqShift;
  int nkShift = 0; while ((1 << nkShift) < Nk) ++nkShift;

  char* ws = (char*)d_ws;
  size_t off = 0;
  auto carve = [&](size_t bytes) -> void* {
    void* p = ws + off;
    off += (bytes + 255) & ~size_t(255);
    return p;
  };
  bf16_t* Wq_b = (bf16_t*)carve(256 * 256 * sizeof(bf16_t));
  bf16_t* Wk_b = (bf16_t*)carve(256 * 256 * sizeof(bf16_t));
  bf16_t* Wv_b = (bf16_t*)carve(256 * 256 * sizeof(bf16_t));
  bf16_t* Wo_b = (bf16_t*)carve(256 * 256 * sizeof(bf16_t));
  float*  Qf   = (float*) carve((size_t)rowsQ * 256 * sizeof(float));
  bf16_t* Qb   = (bf16_t*)carve((size_t)rowsQ * 256 * sizeof(bf16_t));
  bf16_t* Kb   = (bf16_t*)carve((size_t)rowsK * 256 * sizeof(bf16_t));
  bf16_t* Vt   = (bf16_t*)carve((size_t)rowsK * 256 * sizeof(bf16_t));
  float*  Obuf = (float*) carve((size_t)rowsQ * 256 * sizeof(float));
  float*  X0f  = (float*) carve((size_t)rowsQ * 256 * sizeof(float));
  bf16_t* X0b  = (bf16_t*)carve((size_t)rowsQ * 256 * sizeof(bf16_t));
  float*  Z    = Obuf;  // reuse: Obuf is dead after LN0

  const int wn = 256 * 256;
  cvt_bf16_kernel<<<dim3((wn + 255) / 256), dim3(256), 0, stream>>>(Wq, Wq_b, wn);
  cvt_bf16_kernel<<<dim3((wn + 255) / 256), dim3(256), 0, stream>>>(Wk, Wk_b, wn);
  cvt_bf16_kernel<<<dim3((wn + 255) / 256), dim3(256), 0, stream>>>(Wv, Wv_b, wn);
  cvt_bf16_kernel<<<dim3((wn + 255) / 256), dim3(256), 0, stream>>>(Wo, Wo_b, wn);

  // projections (one wave32 per 16x64 tile)
  proj_kernel<<<dim3(rowsQ / 16, 4), dim3(32), 0, stream>>>(q, Wq_b, bq, Qf, Qb, Nq, nqShift, 0);
  proj_kernel<<<dim3(rowsK / 16, 4), dim3(32), 0, stream>>>(k, Wk_b, bk, nullptr, Kb, Nk, nkShift, 1);
  proj_kernel<<<dim3(rowsK / 16, 4), dim3(32), 0, stream>>>(k, Wv_b, bv, nullptr, Vt, Nk, nkShift, 2);

  // flash attention + per-head residual (4 q-tiles per block share K/V DMA)
  attn_kernel<<<dim3(Nq / 64, 8, B), dim3(128), 0, stream>>>(Qb, Kb, Vt, Qf, Obuf, Nq, Nk);

  // LN0 -> MLP (relu + residual) -> LN1
  ln_kernel<<<dim3((rowsQ + 7) / 8), dim3(256), 0, stream>>>(Obuf, g0, b0, X0f, X0b, rowsQ);
  ffn_kernel<<<dim3(rowsQ / 16, 4), dim3(32), 0, stream>>>(X0b, Wo_b, bo, X0f, Z);
  ln_kernel<<<dim3((rowsQ + 7) / 8), dim3(256), 0, stream>>>(Z, g1, b1, out, nullptr, rowsQ);
}
